// TPUGEMMLinear_9509057593375
// MI455X (gfx1250) — 8-bit matrix operand order repaired and checked by reading, compiled, not run on hardware
//
#include <hip/hip_runtime.h>
#include <hip/hip_bf16.h>

#define K_DIM 1024
#define N_DIM 1024
#define KT_COUNT 8
#define NT_COUNT (N_DIM / 16)
#define TILE_BYTES 2048

typedef __attribute__((ext_vector_type(16))) int   v16i;
typedef __attribute__((ext_vector_type(8)))  float v8f;
typedef __attribute__((ext_vector_type(4)))  int   v4i;

typedef __attribute__((address_space(1))) v4i gv4i;
typedef __attribute__((address_space(3))) v4i lv4i;

union Frag { v16i v; v4i q[4]; };
union Acc8 { v8f  v; float f[8]; };

__device__ inline unsigned int f32_to_e4m3_sw(float f) {
  unsigned int u = __float_as_uint(f);
  unsigned int sign = (u >> 24) & 0x80u;
  float af = fabsf(f);
  if (af != af) return sign | 0x7Fu;
  if (af > 448.0f) return sign | 0x7Eu;
  unsigned int x = __float_as_uint(af);
  int exp = (int)(x >> 23) - 127;
  if (af < 0.015625f) {
    int v = (int)rintf(af * 512.0f);
    if (v > 8) v = 8;
    return sign | (unsigned int)v;
  }
  unsigned int mant = x & 0x7FFFFFu;
  unsigned int m3   = mant >> 20;
  unsigned int rest = mant & 0xFFFFFu;
  unsigned int rnd  = (rest > 0x80000u) || (rest == 0x80000u && (m3 & 1u));
  m3 += rnd;
  if (m3 == 8u) { m3 = 0u; exp += 1; }
  if (exp > 8) return sign | 0x7Eu;
  return sign | (((unsigned int)(exp + 7)) << 3) | m3;
}

__device__ inline unsigned int pack4_e4m3(float a, float b, float c, float d) {
#if __has_builtin(__builtin_amdgcn_cvt_pk_fp8_f32)
  int r = 0;
  r = __builtin_amdgcn_cvt_pk_fp8_f32(a, b, r, false);
  r = __builtin_amdgcn_cvt_pk_fp8_f32(c, d, r, true);
  return (unsigned int)r;
#else
  return f32_to_e4m3_sw(a) | (f32_to_e4m3_sw(b) << 8) |
         (f32_to_e4m3_sw(c) << 16) | (f32_to_e4m3_sw(d) << 24);
#endif
}

__device__ inline float bf16_rne(float x) {
  return __bfloat162float(__float2bfloat16(x));
}

__device__ inline void async_cp16(const void* g, void* l) {
#if __has_builtin(__builtin_amdgcn_global_load_async_to_lds_b128)
  __builtin_amdgcn_global_load_async_to_lds_b128(
      (gv4i*)(g), (lv4i*)(l), 0, 0);
#else
  *(v4i*)l = *(const v4i*)g;
#endif
}

__device__ inline void wait_async0() {
#if __has_builtin(__builtin_amdgcn_s_wait_asynccnt)
  __builtin_amdgcn_s_wait_asynccnt(0);
#else
  asm volatile("s_wait_asynccnt 0" ::: "memory");
#endif
}

__global__ __launch_bounds__(256) void quant_x_kernel(
    const float* __restrict__ x, const float* __restrict__ s_in,
    unsigned char* __restrict__ Aq, int mt_count)
{
  unsigned int g = blockIdx.x * 256u + threadIdx.x;
  unsigned int chunk = g & 7u;
  unsigned int lane  = (g >> 3) & 31u;
  unsigned int kt    = (g >> 8) & 7u;
  unsigned int mt    = g >> 11;
  if (mt >= (unsigned int)mt_count) return;

  unsigned int m  = mt * 16u + (lane & 15u);
  unsigned int kb = kt * 128u + (chunk >> 1) * 32u + ((lane >> 4) << 4) + (chunk & 1u) * 8u;

  const float4* s4 = (const float4*)(x + (size_t)m * K_DIM + kb);
  float4 p0 = s4[0], p1 = s4[1];
  float inv = 1.0f / s_in[0];

  float v0 = bf16_rne(p0.x) * inv, v1 = bf16_rne(p0.y) * inv;
  float v2 = bf16_rne(p0.z) * inv, v3 = bf16_rne(p0.w) * inv;
  float v4 = bf16_rne(p1.x) * inv, v5 = bf16_rne(p1.y) * inv;
  float v6 = bf16_rne(p1.z) * inv, v7 = bf16_rne(p1.w) * inv;

  unsigned int lo = pack4_e4m3(v0, v1, v2, v3);
  unsigned int hi = pack4_e4m3(v4, v5, v6, v7);

  unsigned char* dst = Aq + ((size_t)(kt * mt_count + mt)) * TILE_BYTES
                          + lane * 64u + chunk * 8u;
  *(unsigned long long*)dst = ((unsigned long long)hi << 32) | lo;
}

__global__ __launch_bounds__(256) void quant_w_kernel(
    const __hip_bfloat16* __restrict__ w, const float* __restrict__ s_k,
    unsigned char* __restrict__ Bq)
{
  unsigned int g = blockIdx.x * 256u + threadIdx.x;
  unsigned int chunk = g & 3u;
  unsigned int lane  = (g >> 2) & 31u;
  unsigned int kt    = (g >> 7) & 7u;
  unsigned int nt    = g >> 10;
  if (nt >= NT_COUNT) return;

  unsigned int n  = nt * 16u + (lane & 15u);
  unsigned int kb = kt * 128u + chunk * 32u + ((lane >> 4) << 4);
  float inv = 1.0f / s_k[0];

  unsigned int words[4];
#pragma unroll
  for (int q = 0; q < 4; ++q) {
    float a = __bfloat162float(w[(size_t)(kb + q * 4 + 0) * N_DIM + n]) * inv;
    float b = __bfloat162float(w[(size_t)(kb + q * 4 + 1) * N_DIM + n]) * inv;
    float c = __bfloat162float(w[(size_t)(kb + q * 4 + 2) * N_DIM + n]) * inv;
    float d = __bfloat162float(w[(size_t)(kb + q * 4 + 3) * N_DIM + n]) * inv;
    words[q] = pack4_e4m3(a, b, c, d);
  }
  unsigned int* dst = (unsigned int*)(Bq + ((size_t)(kt * NT_COUNT + nt)) * TILE_BYTES
                                         + lane * 64u + chunk * 16u);
  dst[0] = words[0]; dst[1] = words[1]; dst[2] = words[2]; dst[3] = words[3];
}

__global__ __launch_bounds__(256) void gemm_fp8_kernel(
    const unsigned char* __restrict__ Aq, const unsigned char* __restrict__ Bq,
    const float* __restrict__ s_in, const float* __restrict__ s_k,
    const __hip_bfloat16* __restrict__ bias, __hip_bfloat16* __restrict__ out,
    int mt_count)
{
  __shared__ __align__(16) unsigned char ldsA[2][8 * TILE_BYTES];
  __shared__ __align__(16) unsigned char ldsB[2][8 * TILE_BYTES];

  const unsigned int tid  = threadIdx.x;
  const unsigned int lane = tid & 31u;
  const unsigned int wave = tid >> 5;
  const unsigned int wm   = wave & 3u;
  const unsigned int wn   = wave >> 2;
  const unsigned int wgm  = blockIdx.x;
  const unsigned int wgn  = blockIdx.y;

  v8f zero = {0.f, 0.f, 0.f, 0.f, 0.f, 0.f, 0.f, 0.f};
  v8f acc[2][4];
#pragma unroll
  for (int i = 0; i < 2; ++i)
#pragma unroll
    for (int j = 0; j < 4; ++j) acc[i][j] = zero;

  auto issue = [&](int kt, int buf) {
    const unsigned char* ga = Aq + ((size_t)(kt * mt_count + wgm * 8)) * TILE_BYTES + tid * 64u;
    const unsigned char* gb = Bq + ((size_t)(kt * NT_COUNT + wgn * 8)) * TILE_BYTES + tid * 64u;
    unsigned char* la = &ldsA[buf][tid * 64u];
    unsigned char* lb = &ldsB[buf][tid * 64u];
#pragma unroll
    for (int c = 0; c < 4; ++c) {
      async_cp16(ga + c * 16, la + c * 16);
      async_cp16(gb + c * 16, lb + c * 16);
    }
  };

  auto compute = [&](int buf) {
    Frag a0, a1;
    const v4i* ap0 = (const v4i*)&ldsA[buf][(wm * 2 + 0) * TILE_BYTES + lane * 64u];
    const v4i* ap1 = (const v4i*)&ldsA[buf][(wm * 2 + 1) * TILE_BYTES + lane * 64u];
    a0.q[0] = ap0[0]; a0.q[1] = ap0[1]; a0.q[2] = ap0[2]; a0.q[3] = ap0[3];
    a1.q[0] = ap1[0]; a1.q[1] = ap1[1]; a1.q[2] = ap1[2]; a1.q[3] = ap1[3];

    Frag bc;
    {
      const v4i* bp = (const v4i*)&ldsB[buf][(wn * 4 + 0) * TILE_BYTES + lane * 64u];
      bc.q[0] = bp[0]; bc.q[1] = bp[1]; bc.q[2] = bp[2]; bc.q[3] = bp[3];
    }
#pragma unroll
    for (int j = 0; j < 4; ++j) {
      Frag bn;
      if (j < 3) {
        const v4i* bq = (const v4i*)&ldsB[buf][(wn * 4 + j + 1) * TILE_BYTES + lane * 64u];
        bn.q[0] = bq[0]; bn.q[1] = bq[1]; bn.q[2] = bq[2]; bn.q[3] = bq[3];
      }
      acc[0][j] = __builtin_amdgcn_wmma_f32_16x16x128_fp8_fp8(
          a0.v, bc.v, (short)0, acc[0][j], false, false);
      acc[1][j] = __builtin_amdgcn_wmma_f32_16x16x128_fp8_fp8(
          a1.v, bc.v, (short)0, acc[1][j], false, false);
      if (j < 3) bc = bn;
    }
  };

  issue(0, 0);
  wait_async0();
  __syncthreads();

  for (int kt = 0; kt < KT_COUNT; ++kt) {
    int cur = kt & 1;
    if (kt + 1 < KT_COUNT) issue(kt + 1, cur ^ 1);
    compute(cur);
    wait_async0();
    __syncthreads();
  }

  __hip_bfloat16* obuf = (__hip_bfloat16*)&ldsA[0][0];
  {
    float scale = s_in[0] * s_k[0];
    unsigned int half = (lane >> 4) ? 8u : 0u;
    unsigned int col  = lane & 15u;
#pragma unroll
    for (int i = 0; i < 2; ++i) {
#pragma unroll
      for (int j = 0; j < 4; ++j) {
        Acc8 a8; a8.v = acc[i][j];
        unsigned int nl = wn * 64u + j * 16u + col;
        float bf = __bfloat162float(bias[wgn * 128u + nl]);
#pragma unroll
        for (int r = 0; r < 8; ++r) {
          unsigned int ml = wm * 32u + i * 16u + half + r;
          obuf[ml * 128u + nl] = __float2bfloat16(a8.f[r] * scale + bf);
        }
      }
    }
  }
  __syncthreads();

  {
    const v4i* src = (const v4i*)obuf;
    unsigned int m0 = wgm * 128u, n0 = wgn * 128u;
#pragma unroll
    for (int it = 0; it < 8; ++it) {
      unsigned int idx = it * 256u + tid;
      unsigned int row = idx >> 4;
      unsigned int c16 = idx & 15u;
      v4i val = src[idx];
      *(v4i*)((unsigned char*)out + (((size_t)(m0 + row) * N_DIM + n0) << 1) + c16 * 16u) = val;
    }
  }
}

extern "C" void kernel_launch(void* const* d_in, const int* in_sizes, int n_in,
                              void* d_out, int out_size, void* d_ws, size_t ws_size,
                              hipStream_t stream) {
  const float*          x    = (const float*)d_in[0];
  const __hip_bfloat16* w    = (const __hip_bfloat16*)d_in[1];
  const float*          s_in = (const float*)d_in[2];
  const float*          s_k  = (const float*)d_in[3];
  const __hip_bfloat16* bias = (const __hip_bfloat16*)d_in[4];
  __hip_bfloat16*       out  = (__hip_bfloat16*)d_out;

  const int M = in_sizes[0] / K_DIM;
  const int mt_count = M / 16;

  unsigned char* Aq = (unsigned char*)d_ws;
  unsigned char* Bq = Aq + (size_t)M * K_DIM;

  quant_x_kernel<<<mt_count * 8, 256, 0, stream>>>(x, s_in, Aq, mt_count);

  quant_w_kernel<<<256, 256, 0, stream>>>(w, s_k, Bq);

  dim3 grid(mt_count / 8, N_DIM / 128);
  gemm_fp8_kernel<<<grid, 256, 0, stream>>>(Aq, Bq, s_in, s_k, bias, out, mt_count);
}
